// SDEDesemprego_57784490000615
// MI455X (gfx1250) — matrix operand order repaired and checked by reading, not compiled, not run on hardware
//
#include <hip/hip_runtime.h>
#include <hip/hip_bf16.h>
#include <math.h>

typedef __attribute__((ext_vector_type(16))) _Float16 v16h;
typedef __attribute__((ext_vector_type(8)))  float    v8f;

#define T_STEPS 1024
#define NMC     32768
#define I0C     1
#define WB      8                 // waves per block (wave32)
#define PPW     16                // paths per wave (= WMMA M)
#define PPB     (WB * PPW)        // 128 paths per block
#define TWO_PI  6.283185307179586f

static __device__ __forceinline__ float fexp(float x) {
    return __builtin_amdgcn_exp2f(x * 1.4426950408889634f);
}
static __device__ __forceinline__ float flog(float x) {
    return __builtin_amdgcn_logf(x) * 0.6931471805599453f;
}
static __device__ __forceinline__ float fsigmoid(float x) {
#if __has_builtin(__builtin_amdgcn_tanhf)
    // sigmoid(x) = 0.5*tanh(x/2) + 0.5  -> single TRANS op (v_tanh_f32)
    return fmaf(0.5f, __builtin_amdgcn_tanhf(x * 0.5f), 0.5f);
#else
    return __builtin_amdgcn_rcpf(1.0f + fexp(-x));
#endif
}
static __device__ __forceinline__ float fsilu(float x) {
#if __has_builtin(__builtin_amdgcn_tanhf)
    // silu(x) = x*sigmoid(x) = h*tanh(h) + h, h = 0.5x  -> mul + tanh + fma
    float h = 0.5f * x;
    return fmaf(h, __builtin_amdgcn_tanhf(h), h);
#else
    return x * __builtin_amdgcn_rcpf(1.0f + fexp(-x));
#endif
}
static __device__ __forceinline__ float fsoftplus(float x) {
    return (x > 20.0f) ? x : flog(1.0f + fexp(x));
}

__global__ __launch_bounds__(256) void sde_mc_kernel(
    const float* __restrict__ X_seq, const float* __restrict__ y_seq,
    const float* __restrict__ kappa_p, const float* __restrict__ a_p,
    const float* __restrict__ b0_p, const float* __restrict__ b1_p,
    const float* __restrict__ b2_p, const float* __restrict__ S_p,
    const float* __restrict__ dW1, const float* __restrict__ db1,
    const float* __restrict__ dW2, const float* __restrict__ db2,
    const float* __restrict__ dW3, const float* __restrict__ db3,
    const float* __restrict__ gW1, const float* __restrict__ gb1,
    const float* __restrict__ gW2, const float* __restrict__ gb2,
    const float* __restrict__ gW3, const float* __restrict__ gb3,
    const float* __restrict__ xi, const int* __restrict__ dt_p,
    float* __restrict__ block_sums)
{
    __shared__ __align__(16) float xs[T_STEPS * 4];      // x0,x1,x2,y per step
    __shared__ __align__(16) float baseBuf[WB][2][64];   // per-wave layer1 base vectors

    const int tid    = threadIdx.x;
    const int lane   = tid & 31;
    const int wave   = tid >> 5;
    const int lane15 = lane & 15;
    const int lgrp   = lane >> 4;
    const int gpath  = blockIdx.x * PPB + wave * PPW + lane15;

    // ---- stage covariates + observations into LDS (once) ----
    for (int i = tid; i < T_STEPS; i += 256) {
        xs[i * 4 + 0] = X_seq[(I0C + i) * 3 + 0];
        xs[i * 4 + 1] = X_seq[(I0C + i) * 3 + 1];
        xs[i * 4 + 2] = X_seq[(I0C + i) * 3 + 2];
        xs[i * 4 + 3] = y_seq[I0C + i];
    }
    __syncthreads();

    // ---- scalars ----
    const float kappa = kappa_p[0];
    const float a0 = a_p[0], a1 = a_p[1], a2 = a_p[2];
    const float b0 = b0_p[0], b1s = b1_p[0], b2s = b2_p[0];
    const float Sv = S_p[0];
    const float w_ang = TWO_PI / Sv;
    const float dtf = (float)dt_p[0];
    const float sqdt = sqrtf(dtf);

    // z0 = logit(clip(y_seq[i0-1]))
    float y0 = fminf(fmaxf(y_seq[I0C - 1], 1e-4f), 1.0f - 1e-4f);
    float z = flog(y0 * __builtin_amdgcn_rcpf(1.0f - y0));

    // ---- loop-invariant weights into registers ----
    // Layer-1 z-row in A-fragment order: element e of chunk c holds
    // k = 32c + lgrp*8 + e + (e>=8 ? 8 : 0)
    float w1zd[32], w1zg[32];
#pragma unroll
    for (int c = 0; c < 2; ++c)
#pragma unroll
        for (int e = 0; e < 16; ++e) {
            int k = 32 * c + lgrp * 8 + e + (e >= 8 ? 8 : 0);
            w1zd[c * 16 + e] = dW1[k];          // row 0 (z weight)
            w1zg[c * 16 + e] = gW1[k];
        }

    // Layer-1 base coefficients: this lane produces base[k] for k = 2*lane, 2*lane+1
    float cbd[2][5], cbg[2][5];
#pragma unroll
    for (int j = 0; j < 2; ++j) {
        int k = 2 * lane + j;
        cbd[j][0] = db1[k];       cbg[j][0] = gb1[k];
        cbd[j][1] = dW1[64 + k];  cbg[j][1] = gW1[64 + k];
        cbd[j][2] = dW1[128 + k]; cbg[j][2] = gW1[128 + k];
        cbd[j][3] = dW1[192 + k]; cbg[j][3] = gW1[192 + k];
        cbd[j][4] = dW1[256 + k]; cbg[j][4] = gW1[256 + k];
    }

    // W2 as resident f16 B-fragments: lane holds column n = 16*tile + lane15,
    // element e of chunk c is K = 32c + 8*lgrp + e + (e>=8 ? 8 : 0)
    v16h Bd[4][2], Bg[4][2];
#pragma unroll
    for (int tile = 0; tile < 4; ++tile) {
        int n = 16 * tile + lane15;
#pragma unroll
        for (int c = 0; c < 2; ++c)
#pragma unroll
            for (int e = 0; e < 16; ++e) {
                int k = 32 * c + 8 * lgrp + e + (e >= 8 ? 8 : 0);
                Bd[tile][c][e] = (_Float16)dW2[k * 64 + n];
                Bg[tile][c][e] = (_Float16)gW2[k * 64 + n];
            }
    }

    float bias2d[4], bias2g[4], w3d[4], w3g[4];
#pragma unroll
    for (int tile = 0; tile < 4; ++tile) {
        int n = 16 * tile + lane15;
        bias2d[tile] = db2[n]; bias2g[tile] = gb2[n];
        w3d[tile]    = dW3[n]; w3g[tile]    = gW3[n];
    }
    const float b3d = db3[0], b3g = gb3[0];

    float* myBaseD = &baseBuf[wave][0][0];
    float* myBaseG = &baseBuf[wave][1][0];

    // MLP tail: layer1 (VALU, direct A-layout) -> layer2 (WMMA) -> layer3 (shuffle reduce)
    auto mlp_tail = [&](const float* bl, const float* w1z, const v16h (&B)[4][2],
                        const float* bias2, const float* w3, float b3, float zv) -> float {
        v16h A0, A1;
#pragma unroll
        for (int e = 0; e < 16; ++e) {
            A0[e] = (_Float16)fsilu(fmaf(zv, w1z[e],      bl[e]));
            A1[e] = (_Float16)fsilu(fmaf(zv, w1z[16 + e], bl[16 + e]));
        }
        const v8f zeroC = {};
        float s[8];
#pragma unroll
        for (int r = 0; r < 8; ++r) s[r] = 0.0f;
#pragma unroll
        for (int tile = 0; tile < 4; ++tile) {
            v8f c;
            c = __builtin_amdgcn_wmma_f32_16x16x32_f16(false, A0, false, B[tile][0],
                                                       (short)0, zeroC, false, false);
            c = __builtin_amdgcn_wmma_f32_16x16x32_f16(false, A1, false, B[tile][1],
                                                       (short)0, c, false, false);
#pragma unroll
            for (int r = 0; r < 8; ++r)
                s[r] = fmaf(fsilu(c[r] + bias2[tile]), w3[tile], s[r]);
        }
        // reduce over N within each 16-lane group
#pragma unroll
        for (int r = 0; r < 8; ++r) {
            s[r] += __shfl_xor(s[r], 1);
            s[r] += __shfl_xor(s[r], 2);
            s[r] += __shfl_xor(s[r], 4);
            s[r] += __shfl_xor(s[r], 8);
        }
        // route result for path (lane&15) back to this lane
        float vo = s[0];
#pragma unroll
        for (int r = 1; r < 8; ++r) vo = ((lane & 7) == r) ? s[r] : vo;
        float voth = __shfl_xor(vo, 16);
        bool own = (((lane >> 3) & 1) == (lane >> 4));
        return (own ? vo : voth) + b3;
    };

    float nacc = 0.0f;
    float xiv = xi[(size_t)0 * NMC + gpath];

    for (int t = 0; t < T_STEPS; ++t) {
        // software-pipelined xi load for next step
        int tn = (t + 1 < T_STEPS) ? (t + 1) : t;
        float xiv_next = xi[(size_t)tn * NMC + gpath];

        float4 xv = *(const float4*)&xs[t * 4];
        const float x0 = xv.x, x1 = xv.y, x2 = xv.z, yobs = xv.w;
        const float tf = (float)(I0C + t);

        // cooperative layer-1 base: 2 k-values per lane, exchanged via wave-private LDS
        {
            float2 vd, vg;
            vd.x = fmaf(tf, cbd[0][4], fmaf(x2, cbd[0][3], fmaf(x1, cbd[0][2], fmaf(x0, cbd[0][1], cbd[0][0]))));
            vd.y = fmaf(tf, cbd[1][4], fmaf(x2, cbd[1][3], fmaf(x1, cbd[1][2], fmaf(x0, cbd[1][1], cbd[1][0]))));
            vg.x = fmaf(tf, cbg[0][4], fmaf(x2, cbg[0][3], fmaf(x1, cbg[0][2], fmaf(x0, cbg[0][1], cbg[0][0]))));
            vg.y = fmaf(tf, cbg[1][4], fmaf(x2, cbg[1][3], fmaf(x1, cbg[1][2], fmaf(x0, cbg[1][1], cbg[1][0]))));
            *(float2*)&myBaseD[2 * lane] = vd;
            *(float2*)&myBaseG[2 * lane] = vg;
        }
        __builtin_amdgcn_wave_barrier();   // LDS ops are in-order within a wave

        // stage the 2x32 base values up-front: wide independent window for the scheduler
        float bld[32], blg[32];
#pragma unroll
        for (int c = 0; c < 2; ++c)
#pragma unroll
            for (int h = 0; h < 2; ++h) {
                const float* p = &myBaseD[32 * c + lgrp * 8 + 16 * h];
                float4 lo = *(const float4*)p;
                float4 hi = *(const float4*)(p + 4);
                int b = c * 16 + h * 8;
                bld[b + 0] = lo.x; bld[b + 1] = lo.y; bld[b + 2] = lo.z; bld[b + 3] = lo.w;
                bld[b + 4] = hi.x; bld[b + 5] = hi.y; bld[b + 6] = hi.z; bld[b + 7] = hi.w;
                const float* q = &myBaseG[32 * c + lgrp * 8 + 16 * h];
                float4 lo2 = *(const float4*)q;
                float4 hi2 = *(const float4*)(q + 4);
                blg[b + 0] = lo2.x; blg[b + 1] = lo2.y; blg[b + 2] = lo2.z; blg[b + 3] = lo2.w;
                blg[b + 4] = hi2.x; blg[b + 5] = hi2.y; blg[b + 6] = hi2.z; blg[b + 7] = hi2.w;
            }

        float fres = mlp_tail(bld, w1zd, Bd, bias2d, w3d, b3d, z);
        float graw = mlp_tail(blg, w1zg, Bg, bias2g, w3g, b3g, z);
        float gdif = fsoftplus(graw) + 1e-6f;

        float ang = tf * w_ang;
        float mu = fmaf(x0, a0, fmaf(x1, a1, fmaf(x2, a2, b0)))
                 + b1s * __sinf(ang) + b2s * __cosf(ang);

        float drift = fmaf(kappa, (mu - z), fres);
        z = fmaf(gdif * sqdt, xiv, fmaf(drift, dtf, z));

        float U  = fsigmoid(z);
        float dy = yobs - U;
        nacc = fmaf(dy, dy, nacc);

        xiv = xiv_next;
    }

    // ---- deterministic block reduction (reuse baseBuf) ----
    __syncthreads();
    float* red = &baseBuf[0][0][0];
    red[tid] = nacc;
    __syncthreads();
#pragma unroll
    for (int s2 = 128; s2 > 0; s2 >>= 1) {
        if (tid < s2) red[tid] += red[tid + s2];
        __syncthreads();
    }
    if (tid == 0) block_sums[blockIdx.x] = red[0];
}

__global__ void sde_finalize_kernel(const float* __restrict__ bs,
                                    const float* __restrict__ lsig_p,
                                    float* __restrict__ out, int nblocks)
{
    float s = 0.0f;
    for (int i = 0; i < nblocks; ++i) s += bs[i];
    float ls = lsig_p[0];
    float sg = __expf(ls) + 1e-8f;
    // each path counted twice (duplicated half-waves): sum = 2 * true sum
    out[0] = 0.25f * s / ((float)T_STEPS * (float)NMC * sg * sg) + ls;
}

extern "C" void kernel_launch(void* const* d_in, const int* in_sizes, int n_in,
                              void* d_out, int out_size, void* d_ws, size_t ws_size,
                              hipStream_t stream)
{
    (void)in_sizes; (void)n_in; (void)out_size; (void)ws_size;

    const float* X_seq = (const float*)d_in[0];
    const float* y_seq = (const float*)d_in[1];
    const float* kappa = (const float*)d_in[2];
    const float* a     = (const float*)d_in[3];
    const float* b0    = (const float*)d_in[4];
    const float* b1    = (const float*)d_in[5];
    const float* b2    = (const float*)d_in[6];
    const float* S     = (const float*)d_in[7];
    const float* lsig  = (const float*)d_in[8];
    const float* dW1   = (const float*)d_in[9];
    const float* db1   = (const float*)d_in[10];
    const float* dW2   = (const float*)d_in[11];
    const float* db2   = (const float*)d_in[12];
    const float* dW3   = (const float*)d_in[13];
    const float* db3   = (const float*)d_in[14];
    const float* gW1   = (const float*)d_in[15];
    const float* gb1   = (const float*)d_in[16];
    const float* gW2   = (const float*)d_in[17];
    const float* gb2   = (const float*)d_in[18];
    const float* gW3   = (const float*)d_in[19];
    const float* gb3   = (const float*)d_in[20];
    const float* xi    = (const float*)d_in[21];
    const int*   dt    = (const int*)d_in[23];

    float* block_sums = (float*)d_ws;
    const int nblocks = NMC / PPB;   // 256 blocks * 8 waves * 16 paths = 32768

    sde_mc_kernel<<<dim3(nblocks), dim3(256), 0, stream>>>(
        X_seq, y_seq, kappa, a, b0, b1, b2, S,
        dW1, db1, dW2, db2, dW3, db3,
        gW1, gb1, gW2, gb2, gW3, gb3,
        xi, dt, block_sums);

    sde_finalize_kernel<<<dim3(1), dim3(1), 0, stream>>>(
        block_sums, lsig, (float*)d_out, nblocks);
}
